// GCNNet_21242908246464
// MI455X (gfx1250) — matrix operand order repaired and checked by reading, not compiled, not run on hardware
//
#include <hip/hip_runtime.h>
#include <hip/hip_bf16.h>

typedef unsigned short u16;
typedef __attribute__((ext_vector_type(16))) __bf16 v16bf;
typedef __attribute__((ext_vector_type(8)))  __bf16 v8bf;
typedef __attribute__((ext_vector_type(8)))  float  v8f;
typedef __attribute__((vector_size(16)))     int    vi4;   // pointee type the async builtin wants

typedef __attribute__((address_space(1))) vi4* gvi4p;
typedef __attribute__((address_space(3))) vi4* lvi4p;

#if __has_builtin(__builtin_amdgcn_global_load_async_to_lds_b128) && \
    __has_builtin(__builtin_amdgcn_s_wait_asynccnt)
#define ATHENA_ASYNC 1
#else
#define ATHENA_ASYNC 0
#endif

__device__ __forceinline__ u16 f2bf(float f) {
    unsigned int u = __float_as_uint(f);
    unsigned int r = u + 0x7FFFu + ((u >> 16) & 1u);   // round-to-nearest-even
    return (u16)(r >> 16);
}
__device__ __forceinline__ float bf2f(u16 h) {
    return __uint_as_float(((unsigned int)h) << 16);
}

// ---------------------------------------------------------------------------
// f32 -> bf16 conversion (grid-stride)
// ---------------------------------------------------------------------------
__global__ __launch_bounds__(256) void cvt_f32_bf16(const float* __restrict__ in,
                                                    u16* __restrict__ out, size_t n) {
    size_t i = (size_t)blockIdx.x * 256u + threadIdx.x;
    size_t stride = (size_t)gridDim.x * 256u;
    for (; i < n; i += stride) out[i] = f2bf(in[i]);
}

// ---------------------------------------------------------------------------
// f32 [R x C] -> bf16 transposed [C x R] (32x32 LDS tiles, coalesced both sides)
// ---------------------------------------------------------------------------
__global__ __launch_bounds__(256) void cvt_transpose_f32_bf16(const float* __restrict__ in,
                                                              u16* __restrict__ out,
                                                              int R, int C) {
    __shared__ float tile[32][33];
    const int bc = blockIdx.x * 32;
    const int br = blockIdx.y * 32;
    const int tx = threadIdx.x & 31;
    const int ty = threadIdx.x >> 5;          // 0..7
    #pragma unroll
    for (int i = 0; i < 32; i += 8)
        tile[ty + i][tx] = in[(size_t)(br + ty + i) * C + bc + tx];
    __syncthreads();
    #pragma unroll
    for (int i = 0; i < 32; i += 8)
        out[(size_t)(bc + ty + i) * R + br + tx] = f2bf(tile[tx][ty + i]);
}

// ---------------------------------------------------------------------------
// bf16 WMMA GEMM: C[M x Ncols] = A[M x K] * B^T, f32 accumulate.
// A: [M x K] row-major.  B: [Ncols x K] row-major (K-contiguous).
// Epilogue: EPI=0 -> Cb = bf16(acc)
//           EPI=1 -> Cb = bf16(tanh(acc + bias[col]))
//           EPI=2 -> Cf = sigmoid(acc)  (f32)
// Block tile 256x128, 256 threads = 8 waves (4x2), wave tile 64x64:
//   per wave per K-step: 16 WMMAs vs 16 ds_load_b128 (compute-dense).
// Double-buffered LDS; global->LDS staging via GLOBAL_LOAD_ASYNC_TO_LDS_B128
// (ASYNCcnt, 6 per wave per stage) when available; sync b128 fallback.
// ---------------------------------------------------------------------------
template<int EPI>
__global__ __launch_bounds__(256) void gemm_bf16(const u16* __restrict__ A,
                                                 const u16* __restrict__ B,
                                                 int M, int Ncols, int K,
                                                 u16* __restrict__ Cb,
                                                 float* __restrict__ Cf,
                                                 const float* __restrict__ bias) {
    __shared__ __align__(64) u16 As[2][256 * 32];   // [m][k], K-contiguous (16KB/buf)
    __shared__ __align__(64) u16 Bs[2][128 * 32];   // [n][k], K-contiguous (8KB/buf)

    const int tid  = threadIdx.x;
    const int lane = tid & 31;
    const int wave = tid >> 5;                   // 0..7
    const int wr   = wave >> 1;                  // 0..3 : 64-row slab
    const int wc   = wave & 1;                   // 0..1 : 64-col slab
    const int blockRow = blockIdx.y * 256;
    const int blockCol = blockIdx.x * 128;

    v8f acc[4][4];
    #pragma unroll
    for (int i = 0; i < 4; ++i)
        #pragma unroll
        for (int j = 0; j < 4; ++j) acc[i][j] = (v8f){0,0,0,0,0,0,0,0};

    // staging: A -> each thread one full 32-elem row (64B, 4 x b128)
    //          B -> each thread a 16-elem half row (32B, 2 x b128)
    const int brow = tid >> 1;                   // 0..127
    const int bcol = (tid & 1) << 4;             // 0 or 16

    auto stage = [&](int buf, int k0) {
        const u16* ga = A + (size_t)(blockRow + tid) * K + k0;
        const u16* gb = B + (size_t)(blockCol + brow) * K + k0 + bcol;
        u16* la = &As[buf][tid * 32];
        u16* lb = &Bs[buf][brow * 32 + bcol];
#if ATHENA_ASYNC
        __builtin_amdgcn_global_load_async_to_lds_b128((gvi4p)ga,        (lvi4p)la,        0, 0);
        __builtin_amdgcn_global_load_async_to_lds_b128((gvi4p)(ga + 8),  (lvi4p)(la + 8),  0, 0);
        __builtin_amdgcn_global_load_async_to_lds_b128((gvi4p)(ga + 16), (lvi4p)(la + 16), 0, 0);
        __builtin_amdgcn_global_load_async_to_lds_b128((gvi4p)(ga + 24), (lvi4p)(la + 24), 0, 0);
        __builtin_amdgcn_global_load_async_to_lds_b128((gvi4p)gb,        (lvi4p)lb,        0, 0);
        __builtin_amdgcn_global_load_async_to_lds_b128((gvi4p)(gb + 8),  (lvi4p)(lb + 8),  0, 0);
#else
        uint4 a0 = *reinterpret_cast<const uint4*>(ga);
        uint4 a1 = *reinterpret_cast<const uint4*>(ga + 8);
        uint4 a2 = *reinterpret_cast<const uint4*>(ga + 16);
        uint4 a3 = *reinterpret_cast<const uint4*>(ga + 24);
        uint4 b0 = *reinterpret_cast<const uint4*>(gb);
        uint4 b1 = *reinterpret_cast<const uint4*>(gb + 8);
        *reinterpret_cast<uint4*>(la)      = a0;
        *reinterpret_cast<uint4*>(la + 8)  = a1;
        *reinterpret_cast<uint4*>(la + 16) = a2;
        *reinterpret_cast<uint4*>(la + 24) = a3;
        *reinterpret_cast<uint4*>(lb)      = b0;
        *reinterpret_cast<uint4*>(lb + 8)  = b1;
#endif
    };

    // per-lane fragment coordinates (wave32 bf16, contiguous k = 16h + s, same order for A and B)
    const int ksA = (lane < 16) ? 0 : 16;    // A: K = {ks..ks+15}
    const int mA  = lane & 15;
    const int kbB = (lane < 16) ? 0 : 16;    // B: K = {kb..kb+15}
    const int nB  = lane & 15;

    const int nk = K >> 5;
    stage(0, 0);

    for (int t = 0; t < nk; ++t) {
        const int buf = t & 1;
        if (t + 1 < nk) {
            stage(buf ^ 1, (t + 1) << 5);
            __builtin_prefetch(A + (size_t)(blockRow + tid) * K + ((t + 2) << 5), 0, 0);
#if ATHENA_ASYNC
            __builtin_amdgcn_s_wait_asynccnt(6);   // leave next stage in flight
#endif
        } else {
#if ATHENA_ASYNC
            __builtin_amdgcn_s_wait_asynccnt(0);
#endif
        }
        __syncthreads();

        // ---- load fragments ----
        v16bf afr[4];
        #pragma unroll
        for (int ti = 0; ti < 4; ++ti) {
            const int m = wr * 64 + ti * 16 + mA;
            v8bf lo = *reinterpret_cast<const v8bf*>(&As[buf][m * 32 + ksA]);
            v8bf hi = *reinterpret_cast<const v8bf*>(&As[buf][m * 32 + ksA + 8]);
            #pragma unroll
            for (int i = 0; i < 8; ++i) { afr[ti][i] = lo[i]; afr[ti][i + 8] = hi[i]; }
        }
        v16bf bfr[4];
        #pragma unroll
        for (int tj = 0; tj < 4; ++tj) {
            const int n = wc * 64 + tj * 16 + nB;
            bfr[tj] = *reinterpret_cast<const v16bf*>(&Bs[buf][n * 32 + kbB]);
        }

        // ---- 16 WMMAs ----
        #pragma unroll
        for (int ti = 0; ti < 4; ++ti)
            #pragma unroll
            for (int tj = 0; tj < 4; ++tj)
                acc[ti][tj] = __builtin_amdgcn_wmma_f32_16x16x32_bf16(
                    false, afr[ti], false, bfr[tj], (short)0, acc[ti][tj], false, false);

        __syncthreads();
    }

    // ---- epilogue: C 16x16 f32 layout: VGPR r -> row r (+8 for lanes 16-31) ----
    #pragma unroll
    for (int ti = 0; ti < 4; ++ti) {
        #pragma unroll
        for (int tj = 0; tj < 4; ++tj) {
            const int row0 = blockRow + wr * 64 + ti * 16 + ((lane >> 4) << 3);
            const int col  = blockCol + wc * 64 + tj * 16 + (lane & 15);
            #pragma unroll
            for (int r2 = 0; r2 < 8; ++r2) {
                float v = acc[ti][tj][r2];
                size_t idx = (size_t)(row0 + r2) * Ncols + col;
                if (EPI == 0) {
                    Cb[idx] = f2bf(v);
                } else if (EPI == 1) {
                    Cb[idx] = f2bf(tanhf(v + bias[col]));
                } else {
                    Cf[idx] = 1.0f / (1.0f + __expf(-v));
                }
            }
        }
    }
}

// ---------------------------------------------------------------------------
// Row softmax fused with elementwise U multiply, in place:  S <- U * softmax(S)
// One block per row; row staged in dynamic LDS.
// ---------------------------------------------------------------------------
__global__ __launch_bounds__(256) void softmax_mul_kernel(const float* __restrict__ U,
                                                          u16* __restrict__ S, int C) {
    extern __shared__ float srow[];
    __shared__ float red[256];
    const int r = blockIdx.x;
    const int t = threadIdx.x;
    const size_t base = (size_t)r * C;

    float lmax = -3.0e38f;
    for (int c = t; c < C; c += 256) {
        float v = bf2f(S[base + c]);
        srow[c] = v;
        lmax = fmaxf(lmax, v);
    }
    red[t] = lmax; __syncthreads();
    #pragma unroll
    for (int s = 128; s > 0; s >>= 1) { if (t < s) red[t] = fmaxf(red[t], red[t + s]); __syncthreads(); }
    const float m = red[0]; __syncthreads();

    float lsum = 0.0f;
    for (int c = t; c < C; c += 256) {
        float e = __expf(srow[c] - m);
        srow[c] = e;
        lsum += e;
    }
    red[t] = lsum; __syncthreads();
    #pragma unroll
    for (int s = 128; s > 0; s >>= 1) { if (t < s) red[t] += red[t + s]; __syncthreads(); }
    const float inv = 1.0f / red[0];

    for (int c = t; c < C; c += 256)
        S[base + c] = f2bf(U[base + c] * srow[c] * inv);
}

// ---------------------------------------------------------------------------
// Row softmax writing f32 (final output Z); bf16 transposed copy made later.
// ---------------------------------------------------------------------------
__global__ __launch_bounds__(256) void softmax_z_kernel(const u16* __restrict__ L,
                                                        float* __restrict__ Zf, int C) {
    extern __shared__ float srow[];
    __shared__ float red[256];
    const int r = blockIdx.x;
    const int t = threadIdx.x;
    const size_t base = (size_t)r * C;

    float lmax = -3.0e38f;
    for (int c = t; c < C; c += 256) {
        float v = bf2f(L[base + c]);
        srow[c] = v;
        lmax = fmaxf(lmax, v);
    }
    red[t] = lmax; __syncthreads();
    #pragma unroll
    for (int s = 128; s > 0; s >>= 1) { if (t < s) red[t] = fmaxf(red[t], red[t + s]); __syncthreads(); }
    const float m = red[0]; __syncthreads();

    float lsum = 0.0f;
    for (int c = t; c < C; c += 256) {
        float e = __expf(srow[c] - m);
        srow[c] = e;
        lsum += e;
    }
    red[t] = lsum; __syncthreads();
    #pragma unroll
    for (int s = 128; s > 0; s >>= 1) { if (t < s) red[t] += red[t + s]; __syncthreads(); }
    const float inv = 1.0f / red[0];

    for (int c = t; c < C; c += 256)
        Zf[base + c] = srow[c] * inv;
}

// ---------------------------------------------------------------------------
extern "C" void kernel_launch(void* const* d_in, const int* in_sizes, int n_in,
                              void* d_out, int out_size, void* d_ws, size_t ws_size,
                              hipStream_t stream) {
    const int Nn = 8192, Dd = 512;
    const float* U  = (const float*)d_in[0];
    const float* X  = (const float*)d_in[1];
    const float* H1 = (const float*)d_in[2];
    const float* W0 = (const float*)d_in[3];
    const float* W1 = (const float*)d_in[4];
    const float* Wa = (const float*)d_in[5];
    const float* ba = (const float*)d_in[6];

    float* Xp = (float*)d_out;                       // [N x N]
    float* Zf = (float*)d_out + (size_t)Nn * Nn;     // [N x D]

    const size_t nn = (size_t)Nn * Nn;
    const size_t nd = (size_t)Nn * Dd;

    char* ws = (char*)d_ws;
    size_t off = 0;
    auto carve = [&](size_t elems) {
        u16* p = (u16*)(ws + off);
        off += ((elems * 2 + 255) & ~(size_t)255);
        return p;
    };
    u16* Ub  = carve(nn);   // bf16(U)            [N x N]
    u16* XT  = carve(nn);   // bf16(X^T)          [N x N]  (K-major for A2@X)
    u16* Sb  = carve(nn);   // S logits -> A2     [N x N]  (in place)
    u16* Tb  = carve(nn);   // A2 @ X             [N x N]
    u16* Kb  = carve(nd);   // tanh(U Wa^T + ba)  [N x D]
    u16* H1b = carve(nd);   // bf16(H1)           [N x D]  (already K-major)
    u16* Wab = carve(nd);   // bf16(Wa)           [D x N]  (already K-major)
    u16* W0T = carve(nd);   // bf16(W0^T)         [D x N]
    u16* W1T = carve(nd);   // bf16(W1^T)         [N x D]
    u16* ZLb = carve(nd);   // Z logits           [N x D]
    u16* ZT  = carve(nd);   // bf16(Z^T)          [D x N]
    u16* UZb = carve(nd);   // U @ Z              [N x D]

    dim3 blk(256);
    // straight conversions (operand already K-contiguous for its GEMM)
    cvt_f32_bf16<<<4096, blk, 0, stream>>>(U,  Ub,  nn);
    cvt_f32_bf16<<<1024, blk, 0, stream>>>(H1, H1b, nd);
    cvt_f32_bf16<<<1024, blk, 0, stream>>>(Wa, Wab, nd);
    // transposed conversions so every GEMM B-operand is [Ncols x K]
    cvt_transpose_f32_bf16<<<dim3(Nn / 32, Nn / 32), blk, 0, stream>>>(X,  XT,  Nn, Nn);
    cvt_transpose_f32_bf16<<<dim3(Dd / 32, Nn / 32), blk, 0, stream>>>(W0, W0T, Nn, Dd);
    cvt_transpose_f32_bf16<<<dim3(Nn / 32, Dd / 32), blk, 0, stream>>>(W1, W1T, Dd, Nn);

    dim3 gSmall(Dd / 128, Nn / 256);   // N=512  output tiles
    dim3 gBig(Nn / 128, Nn / 256);     // N=8192 output tiles

    // K = tanh(U @ Wa^T + ba)            [N x D], Kdim = N
    gemm_bf16<1><<<gSmall, blk, 0, stream>>>(Ub, Wab, Nn, Dd, Nn, Kb, nullptr, ba);
    // Slogits = K @ H1^T                 [N x N], Kdim = D
    gemm_bf16<0><<<gBig, blk, 0, stream>>>(Kb, H1b, Nn, Nn, Dd, Sb, nullptr, nullptr);
    // A2 = U * softmax(Slogits)          (in place in Sb)
    softmax_mul_kernel<<<Nn, blk, Nn * sizeof(float), stream>>>(U, Sb, Nn);
    // T = A2 @ X                         [N x N], Kdim = N   (dominant N^3 GEMM)
    gemm_bf16<0><<<gBig, blk, 0, stream>>>(Sb, XT, Nn, Nn, Nn, Tb, nullptr, nullptr);
    // Zlogits = T @ W0                   [N x D], Kdim = N
    gemm_bf16<0><<<gSmall, blk, 0, stream>>>(Tb, W0T, Nn, Dd, Nn, ZLb, nullptr, nullptr);
    // Z = softmax(Zlogits) -> f32 output
    softmax_z_kernel<<<Nn, blk, Dd * sizeof(float), stream>>>(ZLb, Zf, Dd);
    // bf16(Z^T) for the next GEMM
    cvt_transpose_f32_bf16<<<dim3(Dd / 32, Nn / 32), blk, 0, stream>>>(Zf, ZT, Nn, Dd);
    // UZ = U @ Z                         [N x D], Kdim = N
    gemm_bf16<0><<<gSmall, blk, 0, stream>>>(Ub, ZT, Nn, Dd, Nn, UZb, nullptr, nullptr);
    // X_p = sigmoid(UZ @ W1)             [N x N], Kdim = D  (f32 -> d_out)
    gemm_bf16<2><<<gBig, blk, 0, stream>>>(UZb, W1T, Nn, Nn, Dd, nullptr, Xp, nullptr);
}
